// mLSTMLayerVision_61744449847872
// MI455X (gfx1250) — matrix operand order repaired and checked by reading, not compiled, not run on hardware
//
#include <hip/hip_runtime.h>
#include <hip/hip_bf16.h>

typedef __attribute__((ext_vector_type(16))) _Float16 v16h;
typedef __attribute__((ext_vector_type(8)))  _Float16 v8h;
typedef __attribute__((ext_vector_type(8)))  float    v8f;

#define B_   4
#define S_   1024
#define D_   768
#define NH_  12
#define QK_  384
#define DQK_ 32
#define DV_  64
#define HW_  32
#define QSCALE 0.17677669529663687f   // 1/sqrt(32)

__device__ __forceinline__ v8f zero8f() {
  v8f z;
#pragma unroll
  for (int i = 0; i < 8; ++i) z[i] = 0.0f;
  return z;
}

// ---- WMMA fragment loaders (layouts per CDNA5 ISA 7.12.2, wave32) ----
// A 16x32 f16, row-major [M,K]: lane m=lane&15, halves 0..7 = K kh..kh+7,
// halves 8..15 = K 16+kh..16+kh+7, kh = (lane<16)?0:8.
__device__ __forceinline__ v16h frag_a_load(const _Float16* base, int lda, int lane) {
  const int m  = lane & 15;
  const int kh = (lane < 16) ? 0 : 8;
  const _Float16* r = base + (size_t)m * lda + kh;
  v8h lo = *(const v8h*)(r);
  v8h hi = *(const v8h*)(r + 16);
  v16h f;
#pragma unroll
  for (int j = 0; j < 8; ++j) { f[j] = lo[j]; f[j + 8] = hi[j]; }
  return f;
}

// B 32x16 f16 from transposed storage Bt[N,K] row-major:
// lane n=lane&15, halves 0..7 = K kb..kb+7, halves 8..15 = K 16+kb..16+kb+7, kb=(lane<16)?0:8.
__device__ __forceinline__ v16h frag_b_load(const _Float16* bt, int ldb, int lane) {
  const int n  = lane & 15;
  const int kb = (lane < 16) ? 0 : 8;
  const _Float16* r = bt + (size_t)n * ldb + kb;
  v8h lo = *(const v8h*)(r);
  v8h hi = *(const v8h*)(r + 16);
  v16h f;
#pragma unroll
  for (int j = 0; j < 8; ++j) { f[j] = lo[j]; f[j + 8] = hi[j]; }
  return f;
}

__device__ __forceinline__ v16h frag_from(v8h lo, v8h hi) {
  v16h f;
#pragma unroll
  for (int j = 0; j < 8; ++j) { f[j] = lo[j]; f[j + 8] = hi[j]; }
  return f;
}

__device__ __forceinline__ v8f wmma16(v16h a, v16h b, v8f c) {
  return __builtin_amdgcn_wmma_f32_16x16x32_f16(false, a, false, b, (short)0, c, false, false);
}

// =============================== pack kernels ===============================
__global__ void k_f32_to_f16(const float* __restrict__ src, _Float16* __restrict__ dst, int n) {
  int i = blockIdx.x * blockDim.x + threadIdx.x;
  if (i < n) dst[i] = (_Float16)src[i];
}

__global__ void k_zero_f16(_Float16* __restrict__ dst, int n) {
  int i = blockIdx.x * blockDim.x + threadIdx.x;
  if (i < n) dst[i] = (_Float16)0.0f;
}

// src [K,N] f32 row-major -> dst [N,K] f16 row-major
__global__ void k_transpose_f16(const float* __restrict__ src, _Float16* __restrict__ dst,
                                int K, int N) {
  int i = blockIdx.x * blockDim.x + threadIdx.x;
  if (i >= K * N) return;
  int k = i / N, n = i % N;
  dst[(size_t)n * K + k] = (_Float16)src[i];
}

// conv_w [O,I,3,3] f32 -> wtap [9][O][I] f16  (tap = kh*3+kw)
__global__ void k_pack_conv(const float* __restrict__ src, _Float16* __restrict__ dst) {
  int i = blockIdx.x * blockDim.x + threadIdx.x;
  if (i >= D_ * D_ * 9) return;
  int tap = i % 9; int rem = i / 9;
  int ic = rem % D_; int oc = rem / D_;
  dst[((size_t)tap * D_ + oc) * D_ + ic] = (_Float16)src[i];
}

// ================= conv3x3 (implicit GEMM) + bias + SiLU -> a_h =============
// one wave: 16 spatial positions (same image row) x 64 out channels.
// Boundary handling: lanes whose source column is out of image bounds read
// from a zeroed dummy row instead -> no per-element selects in the K loop.
__global__ void __launch_bounds__(128, 1)
k_conv_silu(const _Float16* __restrict__ xh, const _Float16* __restrict__ wtap,
            const float* __restrict__ cb, const _Float16* __restrict__ zrow,
            _Float16* __restrict__ ah) {
  const int lane = threadIdx.x & 31;
  int wid = (blockIdx.x * blockDim.x + threadIdx.x) >> 5;
  int oct = wid % (D_ / 64); wid /= (D_ / 64);
  int wt  = wid % 2;         wid /= 2;
  int hr  = wid % HW_;
  int b   = wid / HW_;
  const int w0  = wt * 16;
  const int oc0 = oct * 64;

  v8f acc[4];
#pragma unroll
  for (int j = 0; j < 4; ++j) acc[j] = zero8f();

  const int m   = lane & 15;
  const int kh8 = (lane < 16) ? 0 : 8;

  for (int kh = 0; kh < 3; ++kh) {
    int hs = hr + kh - 1;
    if (hs < 0 || hs >= HW_) continue;
    for (int kw = 0; kw < 3; ++kw) {
      const int tap  = kh * 3 + kw;
      const int wsrc = w0 + m + kw - 1;
      const bool valid = (wsrc >= 0) && (wsrc < HW_);
      const _Float16* arow =
          valid ? (xh + ((size_t)(b * S_ + hs * HW_ + wsrc)) * D_ + kh8) : (zrow + kh8);
      const _Float16* btap = wtap + ((size_t)tap * D_ + oc0) * D_;

      // software-pipelined K loop (24 steps of 32)
      v8h alo = *(const v8h*)(arow);
      v8h ahi = *(const v8h*)(arow + 16);
      v16h bf[4];
#pragma unroll
      for (int j = 0; j < 4; ++j) bf[j] = frag_b_load(btap + (size_t)j * 16 * D_, D_, lane);

#pragma unroll 2
      for (int k0 = 32; k0 < D_; k0 += 32) {
        v8h nalo = *(const v8h*)(arow + k0);
        v8h nahi = *(const v8h*)(arow + k0 + 16);
        v16h nbf[4];
#pragma unroll
        for (int j = 0; j < 4; ++j)
          nbf[j] = frag_b_load(btap + (size_t)j * 16 * D_ + k0, D_, lane);
        // keep next-stage loads issued before current-stage WMMAs
        __builtin_amdgcn_sched_barrier(0);
        v16h af = frag_from(alo, ahi);
#pragma unroll
        for (int j = 0; j < 4; ++j) acc[j] = wmma16(af, bf[j], acc[j]);
        alo = nalo; ahi = nahi;
#pragma unroll
        for (int j = 0; j < 4; ++j) bf[j] = nbf[j];
      }
      v16h af = frag_from(alo, ahi);
#pragma unroll
      for (int j = 0; j < 4; ++j) acc[j] = wmma16(af, bf[j], acc[j]);
    }
  }

  const int nl = lane & 15;
  const int mb = (lane < 16) ? 0 : 8;
#pragma unroll
  for (int j = 0; j < 4; ++j) {
    const int oc = oc0 + 16 * j + nl;
    const float cbv = cb[oc];
#pragma unroll
    for (int r = 0; r < 8; ++r) {
      size_t row = (size_t)(b * S_ + hr * HW_ + w0 + mb + r) * D_;
      float v = acc[j][r] + cbv;
      v = v / (1.0f + __expf(-v));                 // SiLU
      ah[row + oc] = (_Float16)v;
    }
  }
}

// ======================= generic WMMA GEMM: C = A x Bt^T ====================
// A [M,K] f16 row-major, Bt [N,K] f16 row-major. 256 thr = 8 waves, block
// tile 128(M) x 128(N), wave tile 32x64, K software-pipelined (unroll 2).
// Epilogue: 0: f32 row-major (ldc=N)  2: q (scale + [B,NH,S,DQK] f16)
//           3: k ([B,NH,S,DQK] f16)   4: v (vT [B,NH,DV,S] f16)
__global__ void __launch_bounds__(256, 1)
k_gemm(const _Float16* __restrict__ A, const _Float16* __restrict__ Bt,
       void* __restrict__ Cv, int M, int N, int K, int mode) {
  const int lane = threadIdx.x & 31;
  const int wave = threadIdx.x >> 5;
  const int wy = wave >> 1, wx = wave & 1;
  const int m0 = blockIdx.y * 128 + wy * 32;
  const int n0 = blockIdx.x * 128 + wx * 64;

  v8f acc[2][4];
#pragma unroll
  for (int i = 0; i < 2; ++i)
#pragma unroll
    for (int j = 0; j < 4; ++j) acc[i][j] = zero8f();

  const _Float16* Ar0 = A + (size_t)m0 * K;
  const _Float16* Ar1 = A + (size_t)(m0 + 16) * K;

  v16h a0 = frag_a_load(Ar0, K, lane);
  v16h a1 = frag_a_load(Ar1, K, lane);
  v16h bf[4];
#pragma unroll
  for (int j = 0; j < 4; ++j) bf[j] = frag_b_load(Bt + (size_t)(n0 + 16 * j) * K, K, lane);

#pragma unroll 2
  for (int k0 = 32; k0 < K; k0 += 32) {
    // distance-2 prefetch (speculative; dropped if OOB)
    __builtin_prefetch(Ar0 + k0 + 32, 0, 0);
    __builtin_prefetch(Bt + (size_t)n0 * K + k0 + 32, 0, 0);
    v16h na0 = frag_a_load(Ar0 + k0, K, lane);
    v16h na1 = frag_a_load(Ar1 + k0, K, lane);
    v16h nbf[4];
#pragma unroll
    for (int j = 0; j < 4; ++j)
      nbf[j] = frag_b_load(Bt + (size_t)(n0 + 16 * j) * K + k0, K, lane);
    // keep next-stage loads issued before current-stage WMMAs
    __builtin_amdgcn_sched_barrier(0);
#pragma unroll
    for (int j = 0; j < 4; ++j) {
      acc[0][j] = wmma16(a0, bf[j], acc[0][j]);
      acc[1][j] = wmma16(a1, bf[j], acc[1][j]);
    }
    a0 = na0; a1 = na1;
#pragma unroll
    for (int j = 0; j < 4; ++j) bf[j] = nbf[j];
  }
#pragma unroll
  for (int j = 0; j < 4; ++j) {
    acc[0][j] = wmma16(a0, bf[j], acc[0][j]);
    acc[1][j] = wmma16(a1, bf[j], acc[1][j]);
  }

  const int nl = lane & 15;
  const int mb = (lane < 16) ? 0 : 8;
#pragma unroll
  for (int i = 0; i < 2; ++i) {
#pragma unroll
    for (int j = 0; j < 4; ++j) {
#pragma unroll
      for (int r = 0; r < 8; ++r) {
        const int gm = m0 + 16 * i + mb + r;
        const int gn = n0 + 16 * j + nl;
        float val = acc[i][j][r];
        if (mode == 0) {
          ((float*)Cv)[(size_t)gm * N + gn] = val;
        } else if (mode == 2 || mode == 3) {
          if (mode == 2) val *= QSCALE;
          int b = gm >> 10, s = gm & 1023;
          int h = gn >> 5,  dq = gn & 31;
          ((_Float16*)Cv)[(((size_t)(b * NH_ + h)) * S_ + s) * DQK_ + dq] = (_Float16)val;
        } else {  // mode 4: vT [B,NH,DV,S]
          int b = gm >> 10, s = gm & 1023;
          int h = gn >> 6,  dv = gn & 63;
          ((_Float16*)Cv)[(((size_t)(b * NH_ + h)) * DV_ + dv) * S_ + s] = (_Float16)val;
        }
      }
    }
  }
}

// =========================== gates: softcap + logsigmoid ====================
__global__ void k_gates(const float* __restrict__ x, const float* __restrict__ Wi,
                        const float* __restrict__ bi, const float* __restrict__ Wf,
                        const float* __restrict__ bf, float* __restrict__ ig,
                        float* __restrict__ lsf) {
  int id = blockIdx.x * blockDim.x + threadIdx.x;           // (b*S+s)*NH + h
  if (id >= B_ * S_ * NH_) return;
  int h = id % NH_; int bs = id / NH_;
  const float* xr = x + (size_t)bs * D_;
  float ai = 0.f, af = 0.f;
  for (int d = 0; d < D_; ++d) {
    float xv = xr[d];
    ai += xv * Wi[d * NH_ + h];
    af += xv * Wf[d * NH_ + h];
  }
  ai += bi[h]; af += bf[h];
  ai = 15.0f * tanhf(ai * (1.0f / 15.0f));                  // soft cap
  af = 15.0f * tanhf(af * (1.0f / 15.0f));
  float ls = (af >= 0.f) ? -log1pf(__expf(-af)) : (af - log1pf(__expf(af)));
  int b = bs >> 10, s = bs & 1023;
  size_t o = ((size_t)(b * NH_ + h)) * S_ + s;
  ig[o] = ai;
  lsf[o] = ls;
}

__global__ void k_cumsum(const float* __restrict__ lsf, float* __restrict__ lfc) {
  int t = blockIdx.x * blockDim.x + threadIdx.x;
  if (t >= B_ * NH_) return;
  const float* in = lsf + (size_t)t * S_;
  float* out = lfc + (size_t)t * S_;
  float a = 0.f;
  for (int s = 0; s < S_; ++s) { a += in[s]; out[s] = a; }
}

// ==================== flash-style stabilized mLSTM per (b,h) ================
// one wave per (b,h, 16-row tile). online row max m, distributed csum, and
// rescaled WMMA accumulators for C*V (DV=64 -> 4 accumulators). k/v fragments
// hoisted to the top of each 32-col step so exp/shfl VALU hides load latency.
__global__ void __launch_bounds__(32, 1)
k_mlstm(const _Float16* __restrict__ qh, const _Float16* __restrict__ kh,
        const _Float16* __restrict__ vT, const float* __restrict__ lfc,
        const float* __restrict__ ig, float* __restrict__ hws) {
  __shared__ __align__(32) _Float16 ldsC[16 * 32];
  const int lane = threadIdx.x & 31;
  const int bid = blockIdx.x;
  const int st = bid & 63;
  const int bh = bid >> 6;                    // b*NH + h
  const int s0 = st * 16;
  const float NEG_INF = -__builtin_inff();

  const v16h qa = frag_a_load(qh + ((size_t)bh * S_ + s0) * DQK_, DQK_, lane);
  const float* lfc_bh = lfc + (size_t)bh * S_;
  const float* ig_bh  = ig + (size_t)bh * S_;
  const int nl = lane & 15;
  const int mb = (lane < 16) ? 0 : 8;

  float lfs[8], mrun[8], csum[8];
#pragma unroll
  for (int r = 0; r < 8; ++r) {
    lfs[r] = lfc_bh[s0 + mb + r];
    mrun[r] = NEG_INF;
    csum[r] = 0.f;
  }
  v8f acc[4];
#pragma unroll
  for (int j = 0; j < 4; ++j) acc[j] = zero8f();

  auto process_half = [&](v16h kb, int tt0, int half) {
    v8f sc = wmma16(qa, kb, zero8f());
    const int t = tt0 + nl;
    const float lft = lfc_bh[t];
    const float it  = ig_bh[t];
#pragma unroll
    for (int r = 0; r < 8; ++r) {
      const int s = s0 + mb + r;
      float ld = (t <= s) ? (lfs[r] - lft + it) : NEG_INF;
      float rm = ld;
      rm = fmaxf(rm, __shfl_xor(rm, 1, 32));
      rm = fmaxf(rm, __shfl_xor(rm, 2, 32));
      rm = fmaxf(rm, __shfl_xor(rm, 4, 32));
      rm = fmaxf(rm, __shfl_xor(rm, 8, 32));
      float mnew = fmaxf(mrun[r], rm);
      float scl = (mrun[r] == NEG_INF) ? 0.0f : __expf(mrun[r] - mnew);
      float cd  = (ld == NEG_INF) ? 0.0f : __expf(ld - mnew) * sc[r];
      csum[r] = csum[r] * scl + cd;
      acc[0][r] *= scl; acc[1][r] *= scl; acc[2][r] *= scl; acc[3][r] *= scl;
      ldsC[(mb + r) * 32 + 16 * half + nl] = (_Float16)cd;
      mrun[r] = mnew;
    }
  };

  const int tpmax = (s0 + 15) / 32;
  for (int tp = 0; tp <= tpmax; ++tp) {
    const int t0 = tp * 32;
    // hoisted loads: rows t0..t0+31 are always < S (t0+31 <= 1023), so these
    // loads are unconditionally safe; masking happens in the compute only.
    v16h kb0 = frag_b_load(kh + ((size_t)bh * S_ + t0) * DQK_, DQK_, lane);
    v16h kb1 = frag_b_load(kh + ((size_t)bh * S_ + t0 + 16) * DQK_, DQK_, lane);
    v16h vb[4];
#pragma unroll
    for (int j = 0; j < 4; ++j)
      vb[j] = frag_b_load(vT + ((size_t)bh * DV_ + j * 16) * S_ + t0, S_, lane);

    process_half(kb0, t0, 0);
    if (t0 + 16 <= s0 + 15) {
      process_half(kb1, t0 + 16, 1);
    } else {
      // tile fully masked: zero its LDS region
      if (lane < 16) {
        v8h z;
#pragma unroll
        for (int j = 0; j < 8; ++j) z[j] = (_Float16)0.0f;
        *(v8h*)(&ldsC[lane * 32 + 16]) = z;
        *(v8h*)(&ldsC[lane * 32 + 24]) = z;
      }
    }
    asm volatile("s_wait_dscnt 0" ::: "memory");
    v16h ca = frag_a_load(ldsC, 32, lane);
#pragma unroll
    for (int j = 0; j < 4; ++j) acc[j] = wmma16(ca, vb[j], acc[j]);
    asm volatile("" ::: "memory");
  }

#pragma unroll
  for (int r = 0; r < 8; ++r) {
    float cs = csum[r];
    cs += __shfl_xor(cs, 1, 32);
    cs += __shfl_xor(cs, 2, 32);
    cs += __shfl_xor(cs, 4, 32);
    cs += __shfl_xor(cs, 8, 32);
    float n = fmaxf(fabsf(cs), __expf(-mrun[r]));
    float inv = 1.0f / (n + 1e-6f);
    size_t ob = ((size_t)bh * S_ + s0 + mb + r) * DV_;
    hws[ob + 0 * 16 + nl] = acc[0][r] * inv;
    hws[ob + 1 * 16 + nl] = acc[1][r] * inv;
    hws[ob + 2 * 16 + nl] = acc[2][r] * inv;
    hws[ob + 3 * 16 + nl] = acc[3][r] * inv;
  }
}

// ================= per-head LayerNorm + sigmoid output gate -> f16 ==========
__global__ void k_ln_gate(const float* __restrict__ hws, const float* __restrict__ opre,
                          const float* __restrict__ lnw, _Float16* __restrict__ gh) {
  const int lane = threadIdx.x & 31;
  int gw = (blockIdx.x * blockDim.x + threadIdx.x) >> 5;    // (b, s, h)
  if (gw >= B_ * S_ * NH_) return;
  int h = gw % NH_; int s = (gw / NH_) & 1023; int b = gw / (NH_ * S_);
  const float* hp = hws + (((size_t)(b * NH_ + h)) * S_ + s) * DV_;
  float e0 = hp[lane], e1 = hp[lane + 32];
  float sum = e0 + e1;
  sum += __shfl_xor(sum, 1, 32);  sum += __shfl_xor(sum, 2, 32);
  sum += __shfl_xor(sum, 4, 32);  sum += __shfl_xor(sum, 8, 32);
  sum += __shfl_xor(sum, 16, 32);
  float mu = sum * (1.0f / 64.0f);
  float d0 = e0 - mu, d1 = e1 - mu;
  float vs = d0 * d0 + d1 * d1;
  vs += __shfl_xor(vs, 1, 32);  vs += __shfl_xor(vs, 2, 32);
  vs += __shfl_xor(vs, 4, 32);  vs += __shfl_xor(vs, 8, 32);
  vs += __shfl_xor(vs, 16, 32);
  float rs = rsqrtf(vs * (1.0f / 64.0f) + 1e-6f);
  int c0 = h * DV_ + lane, c1 = c0 + 32;
  size_t row = ((size_t)b * S_ + s) * D_;
  float g0 = d0 * rs * lnw[c0] * (1.0f / (1.0f + __expf(-opre[row + c0])));
  float g1 = d1 * rs * lnw[c1] * (1.0f / (1.0f + __expf(-opre[row + c1])));
  gh[row + c0] = (_Float16)g0;
  gh[row + c1] = (_Float16)g1;
}

// ================================ launch ====================================
extern "C" void kernel_launch(void* const* d_in, const int* in_sizes, int n_in,
                              void* d_out, int out_size, void* d_ws, size_t ws_size,
                              hipStream_t stream) {
  (void)in_sizes; (void)n_in; (void)out_size; (void)ws_size;
  const float* x      = (const float*)d_in[0];
  const float* conv_w = (const float*)d_in[1];
  const float* conv_b = (const float*)d_in[2];
  const float* Wq     = (const float*)d_in[3];
  const float* Wk     = (const float*)d_in[4];
  const float* Wv     = (const float*)d_in[5];
  const float* Wog    = (const float*)d_in[6];
  const float* Wi     = (const float*)d_in[7];
  const float* bi     = (const float*)d_in[8];
  const float* Wf     = (const float*)d_in[9];
  const float* bf     = (const float*)d_in[10];
  const float* ln_w   = (const float*)d_in[11];
  const float* Wout   = (const float*)d_in[12];
  float* out = (float*)d_out;

  char* p = (char*)d_ws;
  auto alloc = [&](size_t bytes) -> char* {
    char* r = p;
    p += (bytes + 255) & ~(size_t)255;
    return r;
  };
  const size_t NXD = (size_t)B_ * S_ * D_;                 // 3,145,728
  _Float16* x_h    = (_Float16*)alloc(NXD * 2);
  _Float16* a_h    = (_Float16*)alloc(NXD * 2);
  _Float16* wq_t   = (_Float16*)alloc((size_t)QK_ * D_ * 2);
  _Float16* wk_t   = (_Float16*)alloc((size_t)QK_ * D_ * 2);
  _Float16* wv_t   = (_Float16*)alloc((size_t)D_ * D_ * 2);
  _Float16* wog_t  = (_Float16*)alloc((size_t)D_ * D_ * 2);
  _Float16* wout_t = (_Float16*)alloc((size_t)D_ * D_ * 2);
  _Float16* wconv  = (_Float16*)alloc((size_t)9 * D_ * D_ * 2);
  _Float16* zrow   = (_Float16*)alloc((size_t)D_ * 2);
  _Float16* q_h    = (_Float16*)alloc((size_t)B_ * NH_ * S_ * DQK_ * 2);
  _Float16* k_h    = (_Float16*)alloc((size_t)B_ * NH_ * S_ * DQK_ * 2);
  _Float16* vT     = (_Float16*)alloc((size_t)B_ * NH_ * DV_ * S_ * 2);
  float*    o_pre  = (float*)alloc(NXD * 4);
  float*    ig     = (float*)alloc((size_t)B_ * NH_ * S_ * 4);
  float*    lsf    = (float*)alloc((size_t)B_ * NH_ * S_ * 4);
  float*    lfc    = (float*)alloc((size_t)B_ * NH_ * S_ * 4);
  float*    h_ws   = (float*)alloc((size_t)B_ * NH_ * S_ * DV_ * 4);
  _Float16* g_h    = (_Float16*)alloc(NXD * 2);

  // ---- pack inputs / weights to f16 (transposed [N,K]) ----
  k_f32_to_f16<<<(NXD + 255) / 256, 256, 0, stream>>>(x, x_h, (int)NXD);
  k_zero_f16<<<(D_ + 255) / 256, 256, 0, stream>>>(zrow, D_);
  k_transpose_f16<<<(D_ * QK_ + 255) / 256, 256, 0, stream>>>(Wq, wq_t, D_, QK_);
  k_transpose_f16<<<(D_ * QK_ + 255) / 256, 256, 0, stream>>>(Wk, wk_t, D_, QK_);
  k_transpose_f16<<<(D_ * D_ + 255) / 256, 256, 0, stream>>>(Wv, wv_t, D_, D_);
  k_transpose_f16<<<(D_ * D_ + 255) / 256, 256, 0, stream>>>(Wog, wog_t, D_, D_);
  k_transpose_f16<<<(D_ * D_ + 255) / 256, 256, 0, stream>>>(Wout, wout_t, D_, D_);
  k_pack_conv<<<(9 * D_ * D_ + 255) / 256, 256, 0, stream>>>(conv_w, wconv);

  // ---- conv3x3 + SiLU (implicit WMMA GEMM) ----
  {
    int waves = B_ * HW_ * (HW_ / 16) * (D_ / 64);         // 3072
    k_conv_silu<<<waves * 32 / 128, 128, 0, stream>>>(x_h, wconv, conv_b, zrow, a_h);
  }

  // ---- projections ----
  const int M = B_ * S_;
  k_gemm<<<dim3(QK_ / 128, M / 128), 256, 0, stream>>>(a_h, wq_t, q_h, M, QK_, D_, 2);
  k_gemm<<<dim3(QK_ / 128, M / 128), 256, 0, stream>>>(a_h, wk_t, k_h, M, QK_, D_, 3);
  k_gemm<<<dim3(D_ / 128, M / 128), 256, 0, stream>>>(x_h, wv_t, vT, M, D_, D_, 4);
  k_gemm<<<dim3(D_ / 128, M / 128), 256, 0, stream>>>(x_h, wog_t, o_pre, M, D_, D_, 0);

  // ---- gates ----
  k_gates<<<(B_ * S_ * NH_ + 255) / 256, 256, 0, stream>>>(x, Wi, bi, Wf, bf, ig, lsf);
  k_cumsum<<<1, 64, 0, stream>>>(lsf, lfc);

  // ---- stabilized mLSTM (flash-style) ----
  k_mlstm<<<B_ * NH_ * (S_ / 16), 32, 0, stream>>>(q_h, k_h, vT, lfc, ig, h_ws);

  // ---- per-head LN + sigmoid gate ----
  k_ln_gate<<<(B_ * S_ * NH_ * 32 + 255) / 256, 256, 0, stream>>>(h_ws, o_pre, ln_w, g_h);

  // ---- output projection ----
  k_gemm<<<dim3(D_ / 128, M / 128), 256, 0, stream>>>(g_h, wout_t, out, M, D_, D_, 0);
}
